// MyModel_39333310496877
// MI455X (gfx1250) — matrix operand order repaired and checked by reading, not compiled, not run on hardware
//
#include <hip/hip_runtime.h>

typedef __attribute__((ext_vector_type(16))) _Float16 v16h;
typedef __attribute__((ext_vector_type(8)))  float    v8f;

#define CEPS 1e-5f

__device__ __forceinline__ float act_apply(float v, int act) {
  switch (act) {
    case 1: return v > 0.f ? v : 0.f;                       // relu
    case 2: return 1.f / (1.f + expf(-v));                  // sigmoid
    case 3: return v > 20.f ? v : log1pf(expf(v));          // softplus
    default: return v;
  }
}

// ---------------------------------------------------------------------------
// WMMA GEMM: C[M,N] = act( epilogue( sum_k A[m,k] * Wt[n,k] ) )
//  - A plain row-major [M, lda]  (gather==0)
//  - or A = implicit im2col of NHWC tensor (gather==1), k-order (ic,ky,kx)
//    matching OIHW weight flattening.
//  - Wt row-major [N, K].
//  Epilogue: +bias[n], +resid[m,n], BN(g,b,m,v per n), activation.
//
//  Block: 256 threads = 8 waves arranged 4(M) x 2(N).
//  Block tile 128(M) x 64(N); each wave computes a 32x32 tile as 2x2
//  v_wmma_f32_16x16x32_f16 with A-fragment reuse across N and B-fragment
//  reuse across M.
//
//  LDS tiles are stored pre-swizzled in WMMA fragment order
//  ([group][lane][16 halves], 32B contiguous per lane) so each fragment is a
//  single v16h load (2x ds_load_b128) per the ISA 7.12.2 layouts:
//    A (16x32): lane = m + 16*k[3],  slot = (k&7) | (k[4]<<3)
//    B (32x16): lane = n + 16*k[3],  slot = (k&7) | (k[4]<<3)
// ---------------------------------------------------------------------------
__global__ __launch_bounds__(256) void k_gemm_wmma(
    const float* __restrict__ A, const float* __restrict__ Wt, float* __restrict__ C,
    int M, int N, int K, int lda, int gather,
    int Hin, int Win, int Cin, int KH, int KW, int padH, int padW,
    const float* __restrict__ bias,
    const float* __restrict__ bng, const float* __restrict__ bnb,
    const float* __restrict__ bnm, const float* __restrict__ bnv,
    const float* __restrict__ resid, int act)
{
  __shared__ __align__(16) _Float16 Asw[8][32][16];  // 8 m-groups of 16 rows
  __shared__ __align__(16) _Float16 Bsw[4][32][16];  // 4 n-groups of 16 cols
  const int tid  = threadIdx.x;
  const int row0 = blockIdx.y * 128;
  const int col0 = blockIdx.x * 64;
  const int wave = tid >> 5;
  const int lane = tid & 31;
  const int wm = wave & 3;   // 0..3 -> 32-row sub-tile
  const int wn = wave >> 2;  // 0..1 -> 32-col sub-tile
  v8f acc00 = {0.f,0.f,0.f,0.f,0.f,0.f,0.f,0.f};
  v8f acc01 = acc00, acc10 = acc00, acc11 = acc00;

  for (int k0 = 0; k0 < K; k0 += 32) {
    // ---- stage A tile 128x32 (f32 -> f16), swizzled, zero-padded ----
    for (int i = tid; i < 128 * 32; i += 256) {
      int r = i >> 5, c = i & 31;
      int m = row0 + r, k = k0 + c;
      float v = 0.f;
      if (m < M && k < K) {
        if (!gather) {
          v = A[(size_t)m * lda + k];
        } else {
          int hw = Hin * Win;
          int b  = m / hw;  int p  = m - b * hw;
          int ho = p / Win; int wo = p - ho * Win;
          int khw = KH * KW;
          int ic = k / khw; int r2 = k - ic * khw;
          int ky = r2 / KW; int kx = r2 - ky * KW;
          int hi = ho + ky - padH, wi = wo + kx - padW;
          if (hi >= 0 && hi < Hin && wi >= 0 && wi < Win)
            v = A[(((size_t)b * Hin + hi) * Win + wi) * Cin + ic];
        }
      }
      int mg = r >> 4, mm = r & 15;
      int ls = mm + (((c >> 3) & 1) << 4);
      int js = (c & 7) | (((c >> 4) & 1) << 3);
      Asw[mg][ls][js] = (_Float16)v;
    }
    // ---- stage B tile 32x64: Wt[n, k], swizzled ----
    for (int i = tid; i < 32 * 64; i += 256) {
      int k = i >> 6, c = i & 63;
      int kk = k0 + k, n = col0 + c;
      float v = (kk < K && n < N) ? Wt[(size_t)n * K + kk] : 0.f;
      int ng = c >> 4, nn = c & 15;
      int ls = nn + (((k >> 3) & 1) << 4);
      int js = (k & 7) | (((k >> 4) & 1) << 3);
      Bsw[ng][ls][js] = (_Float16)v;
    }
    __syncthreads();

    // ---- fragments: one v16h (2x ds_load_b128) each ----
    v16h a0 = *(const v16h*)(&Asw[2 * wm + 0][lane][0]);
    v16h a1 = *(const v16h*)(&Asw[2 * wm + 1][lane][0]);
    v16h b0 = *(const v16h*)(&Bsw[2 * wn + 0][lane][0]);
    v16h b1 = *(const v16h*)(&Bsw[2 * wn + 1][lane][0]);

    acc00 = __builtin_amdgcn_wmma_f32_16x16x32_f16(false, a0, false, b0, (short)0, acc00, false, false);
    acc01 = __builtin_amdgcn_wmma_f32_16x16x32_f16(false, a0, false, b1, (short)0, acc01, false, false);
    acc10 = __builtin_amdgcn_wmma_f32_16x16x32_f16(false, a1, false, b0, (short)0, acc10, false, false);
    acc11 = __builtin_amdgcn_wmma_f32_16x16x32_f16(false, a1, false, b1, (short)0, acc11, false, false);
    __syncthreads();
  }

  // ---- epilogue; D layout: n = lane&15, m = r + 8*(lane>=16) ----
  const int nlo = lane & 15;
  const int mhi = 8 * (lane >> 4);
#pragma unroll
  for (int tj = 0; tj < 2; ++tj) {
    int n = col0 + (2 * wn + tj) * 16 + nlo;
    if (n >= N) continue;
    const float bs = bias ? bias[n] : 0.f;
    float s = 1.f, tsh = 0.f;
    const bool hasbn = (bng != nullptr);
    if (hasbn) { s = bng[n] * rsqrtf(bnv[n] + CEPS); tsh = bnb[n] - bnm[n] * s; }
#pragma unroll
    for (int ti = 0; ti < 2; ++ti) {
      const v8f* ap = (ti == 0) ? (tj == 0 ? &acc00 : &acc01)
                                : (tj == 0 ? &acc10 : &acc11);
#pragma unroll
      for (int r = 0; r < 8; ++r) {
        int m = row0 + (2 * wm + ti) * 16 + r + mhi;
        if (m < M) {
          float v = (*ap)[r] + bs;
          if (resid) v += resid[(size_t)m * N + n];
          if (hasbn) v = v * s + tsh;
          C[(size_t)m * N + n] = act_apply(v, act);
        }
      }
    }
  }
}

// ---------------------------------------------------------------------------
// Fused MHA (torch batch_first=False: attention over L=batch axis).
// qkv: [L*Ntok, 3E] rows t = l*Ntok + n; block = (n, h), threads = L.
// ---------------------------------------------------------------------------
template <int HD>
__global__ void k_attn(const float* __restrict__ qkv, float* __restrict__ out,
                       int Ntok, int E, int L)
{
  extern __shared__ float sm[];
  float* Ks = sm;
  float* Vs = sm + (size_t)L * HD;
  const int n = blockIdx.x, h = blockIdx.y;
  const int E3 = 3 * E;
  for (int i = threadIdx.x; i < L * HD; i += blockDim.x) {
    int m = i / HD, d = i - m * HD;
    const float* row = qkv + ((size_t)m * Ntok + n) * E3;
    Ks[i] = row[E + h * HD + d];
    Vs[i] = row[2 * E + h * HD + d];
  }
  __syncthreads();
  const int l = threadIdx.x;
  if (l >= L) return;
  float q[HD];
  const float* qrow = qkv + ((size_t)l * Ntok + n) * E3 + h * HD;
#pragma unroll
  for (int d = 0; d < HD; ++d) q[d] = qrow[d];
  const float scale = rsqrtf((float)HD);
  float mx = -1e30f;
  for (int m = 0; m < L; ++m) {
    float sc = 0.f;
    const float* kr = Ks + m * HD;
#pragma unroll
    for (int d = 0; d < HD; ++d) sc += q[d] * kr[d];
    sc *= scale;
    mx = fmaxf(mx, sc);
  }
  float o[HD];
#pragma unroll
  for (int d = 0; d < HD; ++d) o[d] = 0.f;
  float sum = 0.f;
  for (int m = 0; m < L; ++m) {
    float sc = 0.f;
    const float* kr = Ks + m * HD;
#pragma unroll
    for (int d = 0; d < HD; ++d) sc += q[d] * kr[d];
    float p = expf(sc * scale - mx);
    sum += p;
    const float* vr = Vs + m * HD;
#pragma unroll
    for (int d = 0; d < HD; ++d) o[d] += p * vr[d];
  }
  const float inv = 1.f / sum;
  float* orow = out + ((size_t)l * Ntok + n) * E + h * HD;
#pragma unroll
  for (int d = 0; d < HD; ++d) orow[d] = o[d] * inv;
}

// ---------------------------------------------------------------------------
// Elementwise / normalization / conv helpers (all NHWC)
// ---------------------------------------------------------------------------
__global__ void k_concat_in(const float* __restrict__ x, const float* __restrict__ x1,
                            float* __restrict__ out, int B, int H, int W)
{
  int i = blockIdx.x * blockDim.x + threadIdx.x;
  int tot = B * H * W * 66;
  if (i >= tot) return;
  int c = i % 66; int p = i / 66;
  int w = p % W;  p /= W;
  int h = p % H;  int b = p / H;
  out[i] = (c < 64) ? x[(((size_t)b * 64 + c) * H + h) * W + w]
                    : x1[(((size_t)b * 2 + (c - 64)) * H + h) * W + w];
}

__global__ void k_layernorm(const float* __restrict__ in, float* __restrict__ out,
                            const float* __restrict__ g, const float* __restrict__ b,
                            int rows, int C)
{
  int r = blockIdx.x * blockDim.x + threadIdx.x;
  if (r >= rows) return;
  const float* x = in + (size_t)r * C;
  float mu = 0.f;
  for (int c = 0; c < C; ++c) mu += x[c];
  mu /= C;
  float var = 0.f;
  for (int c = 0; c < C; ++c) { float d = x[c] - mu; var += d * d; }
  var /= C;
  const float inv = rsqrtf(var + CEPS);
  float* y = out + (size_t)r * C;
  for (int c = 0; c < C; ++c) y[c] = (x[c] - mu) * inv * g[c] + b[c];
}

__global__ void k_rmsnorm(const float* __restrict__ in, float* __restrict__ out,
                          const float* __restrict__ g, int rows, int C)
{
  int r = blockIdx.x * blockDim.x + threadIdx.x;
  if (r >= rows) return;
  const float* x = in + (size_t)r * C;
  float ss = 0.f;
  for (int c = 0; c < C; ++c) ss += x[c] * x[c];
  const float inv = rsqrtf(ss / C + CEPS);
  float* y = out + (size_t)r * C;
  for (int c = 0; c < C; ++c) y[c] = x[c] * inv * g[c];
}

__global__ void k_bn_pos(const float* __restrict__ in, float* __restrict__ out,
                         const float* __restrict__ g, const float* __restrict__ b,
                         const float* __restrict__ m, const float* __restrict__ v,
                         const float* __restrict__ pos, int T, int P, int C)
{
  int i = blockIdx.x * blockDim.x + threadIdx.x;
  if (i >= T * C) return;
  int c = i % C; int t = i / C; int p = t % P;
  float s = g[c] * rsqrtf(v[c] + CEPS);
  out[i] = in[i] * s + (b[c] - m[c] * s) + pos[(size_t)p * C + c];
}

__global__ void k_mamba_dwconv(const float* __restrict__ xc, const float* __restrict__ w,
                               const float* __restrict__ bias, float* __restrict__ u,
                               int Bsz, int L, int ED, int ldin)
{
  int i = blockIdx.x * blockDim.x + threadIdx.x;
  if (i >= Bsz * L * ED) return;
  int e = i % ED; int t = i / ED; int l = t % L; int b = t / L;
  float s = bias[e];
#pragma unroll
  for (int j = 0; j < 4; ++j) {
    int lj = l - 3 + j;
    if (lj >= 0) s += w[e * 4 + j] * xc[((size_t)b * L + lj) * ldin + e];
  }
  u[(size_t)t * ED + e] = s / (1.f + expf(-s));   // silu
}

__global__ void k_mamba_scan(const float* __restrict__ delta, const float* __restrict__ u,
                             const float* __restrict__ xdbc, const float* __restrict__ A_log,
                             const float* __restrict__ Dp, float* __restrict__ y,
                             int Bsz, int L, int ED, int dtr, int ldx)
{
  int idx = blockIdx.x * blockDim.x + threadIdx.x;
  if (idx >= Bsz * ED) return;
  int b = idx / ED, e = idx % ED;
  float a[16], h[16];
#pragma unroll
  for (int n = 0; n < 16; ++n) { a[n] = -expf(A_log[e * 16 + n]); h[n] = 0.f; }
  const float Dv = Dp[e];
  for (int l = 0; l < L; ++l) {
    size_t t = (size_t)b * L + l;
    float dl = delta[t * ED + e];
    float ul = u[t * ED + e];
    const float* Brow = xdbc + t * ldx + dtr;
    const float* Crow = Brow + 16;
    float acc = 0.f;
#pragma unroll
    for (int n = 0; n < 16; ++n) {
      h[n] = expf(dl * a[n]) * h[n] + dl * Brow[n] * ul;
      acc += h[n] * Crow[n];
    }
    y[t * ED + e] = acc + Dv * ul;
  }
}

__global__ void k_gate(float* __restrict__ y, const float* __restrict__ z,
                       int T, int ED, int ldz)
{
  int i = blockIdx.x * blockDim.x + threadIdx.x;
  if (i >= T * ED) return;
  int e = i % ED; int t = i / ED;
  float zz = z[(size_t)t * ldz + e];
  y[i] *= zz / (1.f + expf(-zz));
}

__global__ void k_mean_rows(const float* __restrict__ in, float* __restrict__ out,
                            int B, int P, int C)
{
  int i = blockIdx.x * blockDim.x + threadIdx.x;
  if (i >= B * C) return;
  int b = i / C, c = i % C;
  float s = 0.f;
  for (int p = 0; p < P; ++p) s += in[((size_t)b * P + p) * C + c];
  out[i] = s / (float)P;
}

__global__ void k_scale_bc(const float* __restrict__ in, const float* __restrict__ sc,
                           float* __restrict__ out, int B, int P, int C)
{
  int i = blockIdx.x * blockDim.x + threadIdx.x;
  if (i >= B * P * C) return;
  int c = i % C; int b = i / (P * C);
  out[i] = in[i] * sc[(size_t)b * C + c];
}

__global__ void k_mul(const float* a, const float* b, float* out, int n)
{
  int i = blockIdx.x * blockDim.x + threadIdx.x;
  if (i < n) out[i] = a[i] * b[i];
}

__global__ void k_copy(const float* a, float* out, int n)
{
  int i = blockIdx.x * blockDim.x + threadIdx.x;
  if (i < n) out[i] = a[i];
}

__global__ void k_dwconv2d(const float* __restrict__ in, const float* __restrict__ w,
                           const float* __restrict__ bias, float* __restrict__ out,
                           int Bsz, int H, int W, int C, int KH, int KW, int addflag)
{
  int i = blockIdx.x * blockDim.x + threadIdx.x;
  if (i >= Bsz * H * W * C) return;
  int c = i % C; int p = i / C;
  int wd = p % W; p /= W;
  int h = p % H;  int b = p / H;
  const int ph = (KH - 1) / 2, pw = (KW - 1) / 2;
  float s = bias[c];
  for (int ky = 0; ky < KH; ++ky) {
    int hi = h + ky - ph;
    if (hi < 0 || hi >= H) continue;
    for (int kx = 0; kx < KW; ++kx) {
      int wi = wd + kx - pw;
      if (wi < 0 || wi >= W) continue;
      s += w[((size_t)c * KH + ky) * KW + kx] * in[(((size_t)b * H + hi) * W + wi) * C + c];
    }
  }
  if (addflag) out[i] += s; else out[i] = s;
}

__global__ void k_maxpool2(const float* __restrict__ in, float* __restrict__ out,
                           int Bsz, int H, int W, int C)
{
  int Ho = H / 2, Wo = W / 2;
  int i = blockIdx.x * blockDim.x + threadIdx.x;
  if (i >= Bsz * Ho * Wo * C) return;
  int c = i % C; int p = i / C;
  int wo = p % Wo; p /= Wo;
  int ho = p % Ho; int b = p / Ho;
  int h0 = 2 * ho, w0 = 2 * wo;
  float m = in[(((size_t)b * H + h0) * W + w0) * C + c];
  m = fmaxf(m, in[(((size_t)b * H + h0) * W + w0 + 1) * C + c]);
  m = fmaxf(m, in[(((size_t)b * H + h0 + 1) * W + w0) * C + c]);
  m = fmaxf(m, in[(((size_t)b * H + h0 + 1) * W + w0 + 1) * C + c]);
  out[i] = m;
}

__global__ void k_up2x(const float* __restrict__ in, float* __restrict__ out,
                       int Bsz, int H, int W, int C)
{
  int Ho = 2 * H, Wo = 2 * W;
  int i = blockIdx.x * blockDim.x + threadIdx.x;
  if (i >= Bsz * Ho * Wo * C) return;
  int c = i % C; int p = i / C;
  int wo = p % Wo; p /= Wo;
  int ho = p % Ho; int b = p / Ho;
  float fy = (Ho > 1) ? (float)ho * (float)(H - 1) / (float)(Ho - 1) : 0.f;
  float fx = (Wo > 1) ? (float)wo * (float)(W - 1) / (float)(Wo - 1) : 0.f;
  int i0 = (int)floorf(fy); int i1 = (i0 + 1 < H) ? i0 + 1 : H - 1; float wy = fy - (float)i0;
  int j0 = (int)floorf(fx); int j1 = (j0 + 1 < W) ? j0 + 1 : W - 1; float wx = fx - (float)j0;
  const float* base = in + (size_t)b * H * W * C + c;
  float v00 = base[((size_t)i0 * W + j0) * C];
  float v01 = base[((size_t)i0 * W + j1) * C];
  float v10 = base[((size_t)i1 * W + j0) * C];
  float v11 = base[((size_t)i1 * W + j1) * C];
  out[i] = (v00 * (1.f - wx) + v01 * wx) * (1.f - wy) + (v10 * (1.f - wx) + v11 * wx) * wy;
}

__global__ void k_concat_pad(const float* __restrict__ skip, const float* __restrict__ xa,
                             float* __restrict__ out, int Bsz, int Ho, int Wo,
                             int C1, int C2, int Ha, int Wa, int padT, int padL)
{
  int Ct = C1 + C2;
  int i = blockIdx.x * blockDim.x + threadIdx.x;
  if (i >= Bsz * Ho * Wo * Ct) return;
  int c = i % Ct; int p = i / Ct;
  int w = p % Wo; p /= Wo;
  int h = p % Ho; int b = p / Ho;
  if (c < C1) {
    out[i] = skip[(((size_t)b * Ho + h) * Wo + w) * C1 + c];
  } else {
    int ha = h - padT, wa = w - padL, c2 = c - C1;
    out[i] = (ha >= 0 && ha < Ha && wa >= 0 && wa < Wa)
                 ? xa[(((size_t)b * Ha + ha) * Wa + wa) * C2 + c2] : 0.f;
  }
}

// ---------------------------------------------------------------------------
// Host side
// ---------------------------------------------------------------------------
static inline unsigned gct(long long n) { return (unsigned)((n + 255) / 256); }

static void gemm(hipStream_t st, const float* A, const float* Wt, float* C,
                 int M, int N, int K, int lda,
                 const float* bias, const float* resid, int act,
                 const float* bng = nullptr, const float* bnb = nullptr,
                 const float* bnm = nullptr, const float* bnv = nullptr,
                 int gather = 0, int Hin = 0, int Win = 0, int Cin = 0,
                 int KH = 0, int KW = 0, int padH = 0, int padW = 0)
{
  dim3 grid((N + 63) / 64, (M + 127) / 128);
  k_gemm_wmma<<<grid, 256, 0, st>>>(A, Wt, C, M, N, K, lda, gather,
                                    Hin, Win, Cin, KH, KW, padH, padW,
                                    bias, bng, bnb, bnm, bnv, resid, act);
}

static void run_encoder(hipStream_t st, void* const* din,
                        float* x, float* final_out,
                        int T, int Ntok, int Lbat, int E, int FF, int HD, int heads,
                        const int* bases, int nl, int ng_idx, int nb_idx,
                        float* qkvbuf, float* attbuf, float* tmpbuf)
{
  auto F = [&](int i) { return (const float*)din[i]; };
  for (int li = 0; li < nl; ++li) {
    int b = bases[li];
    const float *in_w = F(b), *in_b = F(b + 1), *out_w = F(b + 2), *out_b = F(b + 3);
    const float *w1 = F(b + 4), *b1 = F(b + 5), *w2 = F(b + 6), *b2 = F(b + 7);
    const float *n1g = F(b + 8), *n1b = F(b + 9), *n2g = F(b + 10), *n2b = F(b + 11);
    // qkv
    gemm(st, x, in_w, qkvbuf, T, 3 * E, E, E, in_b, nullptr, 0);
    // attention
    size_t shm = (size_t)2 * Lbat * HD * sizeof(float);
    if (HD == 9)
      k_attn<9><<<dim3(Ntok, heads), Lbat, shm, st>>>(qkvbuf, attbuf, Ntok, E, Lbat);
    else
      k_attn<32><<<dim3(Ntok, heads), Lbat, shm, st>>>(qkvbuf, attbuf, Ntok, E, Lbat);
    // out proj + residual, then LN1
    gemm(st, attbuf, out_w, tmpbuf, T, E, E, E, out_b, x, 0);
    k_layernorm<<<gct(T), 256, 0, st>>>(tmpbuf, x, n1g, n1b, T, E);
    // FFN
    gemm(st, x, w1, qkvbuf, T, FF, E, E, b1, nullptr, 1);
    gemm(st, qkvbuf, w2, tmpbuf, T, E, FF, FF, b2, x, 0);
    k_layernorm<<<gct(T), 256, 0, st>>>(tmpbuf, x, n2g, n2b, T, E);
  }
  k_layernorm<<<gct(T), 256, 0, st>>>(x, final_out, F(ng_idx), F(nb_idx), T, E);
}

extern "C" void kernel_launch(void* const* d_in, const int* in_sizes, int n_in,
                              void* d_out, int out_size, void* d_ws, size_t ws_size,
                              hipStream_t stream)
{
  (void)in_sizes; (void)n_in; (void)out_size;
  auto F = [&](int i) { return (const float*)d_in[i]; };
  hipStream_t st = stream;

  const int B = 128, HH = 13, WW = 13, PTOK = 169, T0 = B * PTOK; // 21632
  const int T6 = B * 36;   // 4608 tokens at 6x6
  const int ED = 144, DTR = 5, LDX = 37;

  // ---- workspace arena (floats) ----
  float* wsf = (float*)d_ws;
  size_t off = 0;
  auto take = [&](size_t n) { float* p = wsf + off; off += n; return p; };
  float* A0 = take(1557504);   // [T0, 72]  main token activation
  float* A1 = take(2768896);   // [T0, 128] xe1 (persists to up2)
  float* A2 = take(1179648);   // [T6, 256] xe2 (persists to up1)
  float* A3 = take(589824);    // [B*9, 512] xe3
  float* Q0 = take(6240000);
  float* Q1 = take(2400000);
  float* Q2 = take(5600000);
  float* Q3 = take(2800000);
  float* Q4 = take(3200000);
  float* Q5 = take(3200000);
  float* SMa = take(33000);
  float* SMb = take(33000);
  float* SMc = take(33000);
  if (ws_size < off * sizeof(float)) return;  // insufficient workspace

  // ---- concat inputs -> NHWC [B,13,13,66] ----
  k_concat_in<<<gct((long long)T0 * 66), 256, 0, st>>>(F(0), F(1), Q0, B, HH, WW);

  // ---- conv0 (1x1, 66->72) + BN0 + ReLU ----
  gemm(st, Q0, F(2), A0, T0, 72, 66, 66, nullptr, nullptr, 1, F(3), F(4), F(5), F(6));

  // ---- 2x Mamba blocks ----
  for (int ml = 0; ml < 2; ++ml) {
    int b = 7 + ml * 10;
    const float *norm = F(b), *in_w = F(b + 1), *conv_w = F(b + 2), *conv_b = F(b + 3);
    const float *x_w = F(b + 4), *dt_w = F(b + 5), *dt_b = F(b + 6);
    const float *A_log = F(b + 7), *Dd = F(b + 8), *out_w = F(b + 9);
    k_rmsnorm<<<gct(T0), 256, 0, st>>>(A0, Q1, norm, T0, 72);
    gemm(st, Q1, in_w, Q0, T0, 288, 72, 72, nullptr, nullptr, 0);   // [xc|z]
    k_mamba_dwconv<<<gct((long long)T0 * ED), 256, 0, st>>>(Q0, conv_w, conv_b, Q2, B, PTOK, ED, 288);
    gemm(st, Q2, x_w, Q3, T0, LDX, ED, ED, nullptr, nullptr, 0);    // [dt|B|C]
    gemm(st, Q3, dt_w, Q4, T0, ED, DTR, LDX, dt_b, nullptr, 3);     // softplus
    k_mamba_scan<<<gct((long long)B * ED), 256, 0, st>>>(Q4, Q2, Q3, A_log, Dd, Q5, B, PTOK, ED, DTR, LDX);
    k_gate<<<gct((long long)T0 * ED), 256, 0, st>>>(Q5, Q0 + ED, T0, ED, 288);
    gemm(st, Q5, out_w, A0, T0, 72, ED, ED, nullptr, A0, 0);        // + residual
  }

  // ---- transblock (E=72) ----
  k_bn_pos<<<gct((long long)T0 * 72), 256, 0, st>>>(A0, Q1, F(27), F(28), F(29), F(30), F(31), T0, PTOK, 72);
  { int bases[1] = {32};
    run_encoder(st, d_in, Q1, Q1, T0, PTOK, B, 72, 144, 9, 8, bases, 1, 44, 45, Q0, Q2, Q4); }

  // ---- channel attention ----
  k_mean_rows<<<gct((long long)B * 72), 256, 0, st>>>(Q1, SMa, B, PTOK, 72);
  gemm(st, SMa, F(46), SMb, B, 9, 72, 72, F(47), nullptr, 1);
  gemm(st, SMb, F(48), SMc, B, 72, 9, 9, F(49), nullptr, 2);
  k_scale_bc<<<gct((long long)T0 * 72), 256, 0, st>>>(Q1, SMc, A0, B, PTOK, 72);

  // ---- inc: double_conv 72->128->128 (3x3, conv-gather GEMM, BN+ReLU fused) ----
  gemm(st, A0, F(50), Q0, T0, 128, 72 * 9, 0, F(51), nullptr, 1, F(52), F(53), F(54), F(55),
       1, HH, WW, 72, 3, 3, 1, 1);
  gemm(st, Q0, F(56), A1, T0, 128, 128 * 9, 0, F(57), nullptr, 1, F(58), F(59), F(60), F(61),
       1, HH, WW, 128, 3, 3, 1, 1);

  // ---- down1: conv 128->256 + BN + ReLU, then maxpool 13->6 ----
  gemm(st, A1, F(62), Q0, T0, 256, 128 * 9, 0, nullptr, nullptr, 1, F(63), F(64), F(65), F(66),
       1, HH, WW, 128, 3, 3, 1, 1);
  k_maxpool2<<<gct((long long)T6 * 256), 256, 0, st>>>(Q0, Q1, B, HH, WW, 256);

  // ---- MSCA ----
  k_dwconv2d<<<gct((long long)T6 * 256), 256, 0, st>>>(Q1, F(72), F(73), Q2, B, 6, 6, 256, 5, 5, 0);
  k_copy<<<gct((long long)T6 * 256), 256, 0, st>>>(Q2, Q5, T6 * 256);
  { int sB[3] = {74, 78, 82}; int kk[3] = {7, 11, 21};
    for (int i = 0; i < 3; ++i) {
      k_dwconv2d<<<gct((long long)T6 * 256), 256, 0, st>>>(Q2, F(sB[i]), F(sB[i] + 1), Q3, B, 6, 6, 256, 1, kk[i], 0);
      k_dwconv2d<<<gct((long long)T6 * 256), 256, 0, st>>>(Q3, F(sB[i] + 2), F(sB[i] + 3), Q5, B, 6, 6, 256, kk[i], 1, 1);
    } }
  gemm(st, Q5, F(86), Q0, T6, 256, 256, 256, F(87), nullptr, 0);
  k_mul<<<gct((long long)T6 * 256), 256, 0, st>>>(Q0, Q1, A2, T6 * 256);

  // ---- tb1 transblock (E=256, 3 layers) ----
  k_bn_pos<<<gct((long long)T6 * 256), 256, 0, st>>>(A2, Q1, F(88), F(89), F(90), F(91), F(92), T6, 36, 256);
  { int bases[3] = {93, 105, 117};
    run_encoder(st, d_in, Q1, A2, T6, 36, B, 256, 512, 32, 8, bases, 3, 129, 130, Q0, Q2, Q4); }

  // ---- down2: conv 256->512 + BN + ReLU, maxpool 6->3 ----
  gemm(st, A2, F(67), Q0, T6, 512, 256 * 9, 0, nullptr, nullptr, 1, F(68), F(69), F(70), F(71),
       1, 6, 6, 256, 3, 3, 1, 1);
  k_maxpool2<<<gct((long long)B * 9 * 512), 256, 0, st>>>(Q0, A3, B, 6, 6, 512);

  // ---- up1 ----
  k_up2x<<<gct((long long)T6 * 512), 256, 0, st>>>(A3, Q0, B, 3, 3, 512);
  gemm(st, Q0, F(131), Q1, T6, 256, 512, 512, F(132), nullptr, 0);
  k_concat_pad<<<gct((long long)T6 * 512), 256, 0, st>>>(A2, Q1, Q2, B, 6, 6, 256, 256, 6, 6, 0, 0);
  gemm(st, Q2, F(133), Q3, T6, 256, 512 * 9, 0, F(134), nullptr, 1, F(135), F(136), F(137), F(138),
       1, 6, 6, 512, 3, 3, 1, 1);
  gemm(st, Q3, F(139), Q4, T6, 256, 256 * 9, 0, F(140), nullptr, 1, F(141), F(142), F(143), F(144),
       1, 6, 6, 256, 3, 3, 1, 1);

  // ---- up2 ----
  k_up2x<<<gct((long long)B * 144 * 256), 256, 0, st>>>(Q4, Q0, B, 6, 6, 256);
  gemm(st, Q0, F(145), Q1, B * 144, 128, 256, 256, F(146), nullptr, 0);
  k_concat_pad<<<gct((long long)T0 * 256), 256, 0, st>>>(A1, Q1, Q2, B, HH, WW, 128, 128, 12, 12, 0, 0);
  gemm(st, Q2, F(147), Q3, T0, 128, 256 * 9, 0, F(148), nullptr, 1, F(149), F(150), F(151), F(152),
       1, HH, WW, 256, 3, 3, 1, 1);
  gemm(st, Q3, F(153), Q4, T0, 128, 128 * 9, 0, F(154), nullptr, 1, F(155), F(156), F(157), F(158),
       1, HH, WW, 128, 3, 3, 1, 1);

  // ---- classifier ----
  k_mean_rows<<<gct((long long)B * 128), 256, 0, st>>>(Q4, SMa, B, PTOK, 128);
  k_layernorm<<<gct(B), 256, 0, st>>>(SMa, SMb, F(159), F(160), B, 128);
  gemm(st, SMb, F(161), (float*)d_out, B, 11, 128, 128, F(162), nullptr, 0);
}
